// HRWKV_7_50251117363566
// MI455X (gfx1250) — matrix operand order repaired and checked by reading, not compiled, not run on hardware
//
#include <hip/hip_runtime.h>
#include <hip/hip_bf16.h>
#include <math.h>

#define B_   2
#define T_   1024
#define D_   1024
#define H_   16
#define N_   64
#define KVH_ 4
#define M_   (B_*T_)
#define EPS_ 1e-6f

typedef __attribute__((ext_vector_type(16))) __bf16 v16bf;
typedef __attribute__((ext_vector_type(8)))  float  v8f;
typedef __attribute__((__vector_size__(4 * sizeof(int)))) int v4i_;

#if defined(__HIP_DEVICE_COMPILE__) && __has_builtin(__builtin_amdgcn_global_load_async_to_lds_b128)
#define GEMM_ASYNC 1
#else
#define GEMM_ASYNC 0
#endif

union BFU { unsigned short u; __bf16 b; };

static __device__ inline __bf16 f2bf(float f) {
    unsigned u = __float_as_uint(f);
    u = u + 0x7FFFu + ((u >> 16) & 1u);   // round-to-nearest-even
    BFU t; t.u = (unsigned short)(u >> 16);
    return t.b;
}

static __device__ inline float sigm(float x) { return 1.f / (1.f + expf(-x)); }

// 16-byte global -> LDS copy: async DMA on gfx1250, plain store fallback
static __device__ inline void cp16(const float* g, float* l) {
#if GEMM_ASYNC
    __builtin_amdgcn_global_load_async_to_lds_b128(
        (__attribute__((address_space(1))) v4i_*)const_cast<float*>(g),
        (__attribute__((address_space(3))) v4i_*)l, 0, 0);
#else
    *(float4*)l = *(const float4*)g;
#endif
}

static __device__ inline void async_wait0() {
#if GEMM_ASYNC
#if __has_builtin(__builtin_amdgcn_s_wait_asynccnt)
    __builtin_amdgcn_s_wait_asynccnt(0);
#else
    asm volatile("s_wait_asynccnt 0x0" ::: "memory");
#endif
#endif
}

// ---- reductions -----------------------------------------------------------
static __device__ inline float sum64(float v, volatile float* sc) {
    for (int off = 16; off > 0; off >>= 1) v += __shfl_down(v, off, 32);
    if ((threadIdx.x & 31) == 0) sc[threadIdx.x >> 5] = v;
    __syncthreads();
    float t = sc[0] + sc[1];
    __syncthreads();
    return t;
}

static __device__ inline float block_sum256(float v, volatile float* sc) {
    for (int off = 16; off > 0; off >>= 1) v += __shfl_down(v, off, 32);
    if ((threadIdx.x & 31) == 0) sc[threadIdx.x >> 5] = v;
    __syncthreads();
    float t = 0.f;
#pragma unroll
    for (int i = 0; i < 8; i++) t += sc[i];
    __syncthreads();
    return t;
}

// ---- generic bf16-WMMA GEMM: C[M,Nc] = act(A[M,K]) @ W[K,Nc] (+epilogue) ---
// mode: 0 = plain, 1 = +bias[col], 2 = sigmoid(+bias[col])
// actin: 0 = none, 1 = tanh, 2 = sigmoid (applied to A at fragment build)
// Double-buffered tiles, async global->LDS copies (ASYNCcnt handshake).
__global__ __launch_bounds__(256) void gemm_bf16_wmma(
    const float* __restrict__ A, int lda,
    const float* __restrict__ W, int ldw,
    float* __restrict__ C, int ldc,
    int K, const float* __restrict__ bias, int mode, int actin)
{
    __shared__ __align__(16) float sAf[2][128][36]; // 128x32 A tile (pad->16B rows)
    __shared__ __align__(16) float sBf[2][32][68];  // 32x64 B tile, k-major

    const int tid  = threadIdx.x;
    const int lane = tid & 31, wave = tid >> 5;
    const int wm = (wave >> 1) * 32, wn = (wave & 1) * 32;   // 4x2 wave grid
    const int rowBase = blockIdx.y * 128;
    const int colBase = blockIdx.x * 64;
    const int r15 = lane & 15, kh = lane >> 4;

    v8f acc[2][2];
#pragma unroll
    for (int i = 0; i < 2; i++)
#pragma unroll
        for (int j = 0; j < 2; j++)
#pragma unroll
            for (int e = 0; e < 8; e++) acc[i][j][e] = 0.f;

    const int ar = tid >> 1;           // A: 2 threads/row, 16 cols each
    const int ac = (tid & 1) * 16;
    const int bk = tid >> 3;           // B: 8 threads/row, 8 cols each
    const int bn = (tid & 7) * 8;

    auto issue = [&](int kt, int bufi) {
        const float* ga = A + (size_t)(rowBase + ar) * lda + kt * 32 + ac;
        float* la = &sAf[bufi][ar][ac];
#pragma unroll
        for (int j = 0; j < 4; j++) cp16(ga + 4 * j, la + 4 * j);
        const float* gb = W + (size_t)(kt * 32 + bk) * ldw + colBase + bn;
        float* lb = &sBf[bufi][bk][bn];
#pragma unroll
        for (int j = 0; j < 2; j++) cp16(gb + 4 * j, lb + 4 * j);
    };

    const int nk = K >> 5;
    issue(0, 0);
    for (int kt = 0; kt < nk; ++kt) {
        async_wait0();
        __syncthreads();           // tile[cur] complete & visible to all waves
        const int cur = kt & 1;
        if (kt + 1 < nk) issue(kt + 1, cur ^ 1);  // prefetch into other buffer

        const float (*tA)[36] = sAf[cur];
        const float (*tB)[68] = sBf[cur];

        v16bf af[2], bf_[2];
#pragma unroll
        for (int mi = 0; mi < 2; mi++) {
            int mrow = wm + mi * 16 + r15;
#pragma unroll
            for (int i = 0; i < 16; i++) {
                int kk = (i < 8 ? i : i + 8) + kh * 8;   // ISA 16-bit A layout
                float v = tA[mrow][kk];
                if (actin == 1)      v = tanhf(v);
                else if (actin == 2) v = sigm(v);
                af[mi][i] = f2bf(v);
            }
        }
#pragma unroll
        for (int ni = 0; ni < 2; ni++) {
            int ncol = wn + ni * 16 + r15;
#pragma unroll
            for (int i = 0; i < 16; i++)
                bf_[ni][i] = f2bf(tB[(i < 8 ? i : i + 8) + kh * 8][ncol]); // ISA 16-bit B layout
        }
#pragma unroll
        for (int mi = 0; mi < 2; mi++)
#pragma unroll
            for (int ni = 0; ni < 2; ni++)
                acc[mi][ni] = __builtin_amdgcn_wmma_f32_16x16x32_bf16(
                    false, af[mi], false, bf_[ni], (short)0, acc[mi][ni], false, false);
    }

    // C store per ISA 32-bit C/D layout: VGPR j -> M = j (+8 for lanes 16-31)
#pragma unroll
    for (int mi = 0; mi < 2; mi++)
#pragma unroll
        for (int ni = 0; ni < 2; ni++)
#pragma unroll
            for (int j = 0; j < 8; j++) {
                int mrow = rowBase + wm + mi * 16 + (kh ? j + 8 : j);
                int ncol = colBase + wn + ni * 16 + r15;
                float v = acc[mi][ni][j];
                if (mode == 1)      v += bias[ncol];
                else if (mode == 2) v = sigm(v + bias[ncol]);
                C[(size_t)mrow * ldc + ncol] = v;
            }
}

// ---- rms_norm(x_in, ln1) -> X ---------------------------------------------
__global__ __launch_bounds__(256) void rmsnorm_kernel(
    const float* __restrict__ xin, const float* __restrict__ g, float* __restrict__ out)
{
    __shared__ float sc[8];
    size_t row = blockIdx.x;
    int d = threadIdx.x * 4;
    float4 v = *(const float4*)(xin + row * D_ + d);
    float ss = v.x * v.x + v.y * v.y + v.z * v.z + v.w * v.w;
    ss = block_sum256(ss, sc);
    float s = rsqrtf(ss / (float)D_ + EPS_);
    float4 gv = *(const float4*)(g + d);
    float4 o = make_float4(gv.x * v.x * s, gv.y * v.y * s, gv.z * v.z * s, gv.w * v.w * s);
    *(float4*)(out + row * D_ + d) = o;
}

// ---- per-row prep: r/k norms + rope, v/k mixing, decay, kk, buffers -------
__global__ __launch_bounds__(64) void prep_kernel(
    const float* __restrict__ RKVb, const float* __restrict__ Wb,
    const float* __restrict__ Ab,   const float* __restrict__ Vm,
    const float* __restrict__ Km,   const float* __restrict__ cosb,
    const float* __restrict__ sinb, const float* __restrict__ vfirst,
    const float* __restrict__ kfirst,
    const float* __restrict__ Rbias, const float* __restrict__ Kbias,
    const float* __restrict__ Vbias,
    const float* __restrict__ lnr,  const float* __restrict__ lnk,
    float* __restrict__ Rb, float* __restrict__ Db, float* __restrict__ Kf,
    float* __restrict__ Vr, float* __restrict__ Aa, float* __restrict__ Bb)
{
    __shared__ float sc[2];
    __shared__ float rbuf[64];
    __shared__ float kbuf[4][64];
    __shared__ float vbuf[4][64];
    __shared__ float nrm[4];

    size_t row = blockIdx.x;
    int n = threadIdx.x;
    size_t rk = row * 1536;
    float cv = cosb[row * 64 + n];
    float sv = sinb[row * 64 + n];

    // r: 16 heads, rms_norm(ln_r) + rope
    for (int h = 0; h < 16; h++) {
        float x = RKVb[rk + h * 64 + n] + Rbias[h * 64 + n];
        float ms = sum64(x * x, sc) / 64.f;
        float rn = lnr[n] * x * rsqrtf(ms + EPS_);
        rbuf[n] = rn; __syncthreads();
        float rot = (n < 32) ? -rbuf[n + 32] : rbuf[n - 32];
        __syncthreads();
        Rb[row * 1024 + h * 64 + n] = rn * cv + rot * sv;
    }

    // k/v: 4 kv-heads, norm + rope + first-token mixing
    for (int kv = 0; kv < 4; kv++) {
        float x = RKVb[rk + 1024 + kv * 64 + n] + Kbias[kv * 64 + n];
        float ms = sum64(x * x, sc) / 64.f;
        float kn = lnk[n] * x * rsqrtf(ms + EPS_);
        rbuf[n] = kn; __syncthreads();
        float rot = (n < 32) ? -rbuf[n + 32] : rbuf[n - 32];
        __syncthreads();
        float kr = kn * cv + rot * sv;
        float km = Km[row * 256 + kv * 64 + n];
        float kf = kfirst[(row * 4 + kv) * 64 + n];
        float kx = kr + (kf - kr) * km;
        kbuf[kv][n] = kx;
        float s2 = sum64(kx * kx, sc);
        if (n == 0) nrm[kv] = fmaxf(sqrtf(s2), 1e-12f);
        float vraw = RKVb[rk + 1280 + kv * 64 + n] + Vbias[kv * 64 + n];
        float vm = Vm[row * 256 + kv * 64 + n];
        float vf = vfirst[(row * 4 + kv) * 64 + n];
        vbuf[kv][n] = vraw + (vf - vraw) * vm;
    }
    __syncthreads();

    // per-d: decay, scaled k, kk-based aa/bb
    for (int h = 0; h < 16; h++) {
        size_t idx = row * 1024 + h * 64 + n;
        float kr = kbuf[h >> 2][n];
        float a = Ab[idx];
        float wraw = Wb[idx];
        float w = -log1pf(expf(-wraw)) - 0.5f;   // -softplus(-w)-0.5
        float kk = kr / nrm[h >> 2];
        Db[idx] = expf(-expf(w));
        Kf[idx] = kr * (1.f - w + a);
        Vr[idx] = vbuf[h >> 2][n];
        Aa[idx] = -kk;
        Bb[idx] = kk * a;
    }
}

// ---- sequential RWKV-7 state scan: one workgroup per (b,h) ----------------
__global__ __launch_bounds__(256) void scan_kernel(
    const float* __restrict__ Rb, const float* __restrict__ Db,
    const float* __restrict__ Kf, const float* __restrict__ Vr,
    const float* __restrict__ Aa, const float* __restrict__ Bb,
    const float* __restrict__ S0, float* __restrict__ Ob, float* __restrict__ Sfin)
{
    int bh = blockIdx.x;
    int b = bh >> 4, h = bh & 15;
    int tid = threadIdx.x;
    int v = tid >> 2;              // state row 0..63
    int kq = (tid & 3) * 16;       // state col group

    float s[16];
    size_t soff = (((size_t)bh) * 64 + v) * 64 + kq;
#pragma unroll
    for (int j = 0; j < 4; j++) *(float4*)&s[j * 4] = *(const float4*)(S0 + soff + j * 4);

    for (int t = 0; t < T_; t++) {
        size_t base = ((size_t)(b * T_ + t)) * 1024 + (size_t)h * 64;
        float dv[16], av[16], bv[16], kv_[16], rv[16];
#pragma unroll
        for (int j = 0; j < 4; j++) {
            *(float4*)&dv[j * 4]  = *(const float4*)(Db + base + kq + j * 4);
            *(float4*)&av[j * 4]  = *(const float4*)(Aa + base + kq + j * 4);
            *(float4*)&bv[j * 4]  = *(const float4*)(Bb + base + kq + j * 4);
            *(float4*)&kv_[j * 4] = *(const float4*)(Kf + base + kq + j * 4);
            *(float4*)&rv[j * 4]  = *(const float4*)(Rb + base + kq + j * 4);
        }
        float vt = Vr[base + v];

        float pa = 0.f;
#pragma unroll
        for (int j = 0; j < 16; j++) { s[j] *= dv[j]; pa += s[j] * av[j]; }
        pa += __shfl_xor(pa, 1, 32);
        pa += __shfl_xor(pa, 2, 32);   // all 4 lanes of a row get full sa[v]

        float po = 0.f;
#pragma unroll
        for (int j = 0; j < 16; j++) { s[j] += pa * bv[j] + vt * kv_[j]; po += s[j] * rv[j]; }
        po += __shfl_xor(po, 1, 32);
        po += __shfl_xor(po, 2, 32);
        if ((tid & 3) == 0) Ob[base + v] = po;

        if (t + 1 < T_) __builtin_prefetch(Db + base + 1024, 0, 0);  // global_prefetch_b8
    }
#pragma unroll
    for (int j = 0; j < 4; j++) *(float4*)(Sfin + soff + j * 4) = *(const float4*)&s[j * 4];
}

// ---- xx = o/sqrt(N) + bonus, then gate ------------------------------------
__global__ __launch_bounds__(64) void bonus_gate_kernel(
    const float* __restrict__ Ob, const float* __restrict__ Rb,
    const float* __restrict__ Kf, const float* __restrict__ Vr,
    const float* __restrict__ rk_, const float* __restrict__ Gt,
    float* __restrict__ Xg)
{
    __shared__ float sc[2];
    size_t row = blockIdx.x;
    int n = threadIdx.x;
    for (int h = 0; h < 16; h++) {
        size_t idx = row * 1024 + h * 64 + n;
        float r = Rb[idx], kf = Kf[idx], vf = Vr[idx];
        float s = sum64(r * kf * rk_[h * 64 + n], sc);
        float xx = Ob[idx] * 0.125f + s * vf;   // N^-0.5 = 1/8
        Xg[idx] = xx * Gt[idx];
    }
}

// ---- x_new = x_in + Out; outputs rms_norm(x_new, ln2) and x_new -----------
__global__ __launch_bounds__(256) void residual_norm_kernel(
    const float* __restrict__ xin, const float* __restrict__ Ou,
    const float* __restrict__ ln2, float* __restrict__ out0, float* __restrict__ xnew)
{
    __shared__ float sc[8];
    size_t row = blockIdx.x;
    int d = threadIdx.x * 4;
    float4 xi = *(const float4*)(xin + row * D_ + d);
    float4 ov = *(const float4*)(Ou + row * D_ + d);
    float4 xn = make_float4(xi.x + ov.x, xi.y + ov.y, xi.z + ov.z, xi.w + ov.w);
    float ss = xn.x * xn.x + xn.y * xn.y + xn.z * xn.z + xn.w * xn.w;
    ss = block_sum256(ss, sc);
    float s = rsqrtf(ss / (float)D_ + EPS_);
    float4 g = *(const float4*)(ln2 + d);
    *(float4*)(xnew + row * D_ + d) = xn;
    float4 o = make_float4(g.x * xn.x * s, g.y * xn.y * s, g.z * xn.z * s, g.w * xn.w * s);
    *(float4*)(out0 + row * D_ + d) = o;
}

extern "C" void kernel_launch(void* const* d_in, const int* in_sizes, int n_in,
                              void* d_out, int out_size, void* d_ws, size_t ws_size,
                              hipStream_t stream)
{
    (void)in_sizes; (void)n_in; (void)out_size; (void)ws_size;
    const float* x_in   = (const float*)d_in[0];
    const float* state  = (const float*)d_in[1];
    const float* cosb   = (const float*)d_in[2];
    const float* sinb   = (const float*)d_in[3];
    const float* vfirst = (const float*)d_in[4];
    const float* kfirst = (const float*)d_in[5];
    const float* wavgk1 = (const float*)d_in[6];
    const float* w0     = (const float*)d_in[7];
    const float* w2     = (const float*)d_in[8];
    const float* a0     = (const float*)d_in[9];
    const float* a2     = (const float*)d_in[10];
    const float* v0     = (const float*)d_in[11];
    const float* v2     = (const float*)d_in[12];
    const float* g2     = (const float*)d_in[13];
    const float* k0     = (const float*)d_in[14];
    const float* k2     = (const float*)d_in[15];
    const float* r_k    = (const float*)d_in[16];
    const float* RKVw   = (const float*)d_in[17];
    const float* Ow     = (const float*)d_in[18];
    const float* Rbias  = (const float*)d_in[19];
    const float* Kbias  = (const float*)d_in[20];
    const float* Vbias  = (const float*)d_in[21];
    const float* lnr    = (const float*)d_in[22];
    const float* lnk    = (const float*)d_in[23];
    const float* ln1    = (const float*)d_in[24];
    const float* ln2    = (const float*)d_in[25];
    // d_in[26] = layer_id (== 1 for this problem)

    float* ws = (float*)d_ws;
    size_t off = 0;
    auto alloc = [&](size_t n) { float* p = ws + off; off += n; return p; };
    float* X    = alloc((size_t)M_ * D_);
    float* XW   = alloc((size_t)M_ * 448);
    float* RKVb = alloc((size_t)M_ * 1536);
    float* Wb   = alloc((size_t)M_ * D_);
    float* Ab   = alloc((size_t)M_ * D_);
    float* Gt   = alloc((size_t)M_ * D_);
    float* Vm   = alloc((size_t)M_ * 256);
    float* Km   = alloc((size_t)M_ * 256);
    float* Rb   = alloc((size_t)M_ * D_);
    float* Db   = alloc((size_t)M_ * D_);
    float* Kf   = alloc((size_t)M_ * D_);
    float* Vr   = alloc((size_t)M_ * D_);
    float* Aa   = alloc((size_t)M_ * D_);
    float* Bb   = alloc((size_t)M_ * D_);
    float* Ob   = alloc((size_t)M_ * D_);
    float* Xg   = alloc((size_t)M_ * D_);
    float* Ou   = alloc((size_t)M_ * D_);

    float* out = (float*)d_out;
    float* O1 = out;                                  // rms_norm(x_new, ln2)
    float* O2 = O1 + (size_t)M_ * D_;                 // x[:, -1]
    float* O3 = O2 + (size_t)B_ * D_;                 // S_final
    float* O4 = O3 + (size_t)B_ * H_ * N_ * N_;       // v_first (passthrough)
    float* O5 = O4 + (size_t)B_ * T_ * KVH_ * N_;     // k_first (passthrough)
    float* O6 = O5 + (size_t)B_ * T_ * KVH_ * N_;     // x_new

    (void)hipMemcpyAsync(O4, vfirst, (size_t)B_ * T_ * KVH_ * N_ * sizeof(float),
                         hipMemcpyDeviceToDevice, stream);
    (void)hipMemcpyAsync(O5, kfirst, (size_t)B_ * T_ * KVH_ * N_ * sizeof(float),
                         hipMemcpyDeviceToDevice, stream);

    rmsnorm_kernel<<<M_, 256, 0, stream>>>(x_in, ln1, X);

    (void)hipMemcpyAsync(O2,      X + ((size_t)(0 * T_ + T_ - 1)) * D_, D_ * sizeof(float),
                         hipMemcpyDeviceToDevice, stream);
    (void)hipMemcpyAsync(O2 + D_, X + ((size_t)(1 * T_ + T_ - 1)) * D_, D_ * sizeof(float),
                         hipMemcpyDeviceToDevice, stream);

    dim3 blk(256);
    gemm_bf16_wmma<<<dim3(448 / 64, M_ / 128), blk, 0, stream>>>(X, D_, wavgk1, 448, XW, 448, D_, nullptr, 0, 0);
    gemm_bf16_wmma<<<dim3(1536 / 64, M_ / 128), blk, 0, stream>>>(X, D_, RKVw, 1536, RKVb, 1536, D_, nullptr, 0, 0);
    gemm_bf16_wmma<<<dim3(1024 / 64, M_ / 128), blk, 0, stream>>>(XW + 0,   448, w2, 1024, Wb, 1024, 96,  w0,      1, 1);
    gemm_bf16_wmma<<<dim3(1024 / 64, M_ / 128), blk, 0, stream>>>(XW + 96,  448, a2, 1024, Ab, 1024, 96,  a0,      2, 0);
    gemm_bf16_wmma<<<dim3(256 / 64,  M_ / 128), blk, 0, stream>>>(XW + 192, 448, v2, 256,  Vm, 256,  64,  v0,      2, 0);
    gemm_bf16_wmma<<<dim3(1024 / 64, M_ / 128), blk, 0, stream>>>(XW + 256, 448, g2, 1024, Gt, 1024, 128, nullptr, 0, 2);
    gemm_bf16_wmma<<<dim3(256 / 64,  M_ / 128), blk, 0, stream>>>(XW + 384, 448, k2, 256,  Km, 256,  64,  k0,      2, 0);

    prep_kernel<<<M_, 64, 0, stream>>>(RKVb, Wb, Ab, Vm, Km, cosb, sinb, vfirst, kfirst,
                                       Rbias, Kbias, Vbias, lnr, lnk,
                                       Rb, Db, Kf, Vr, Aa, Bb);

    scan_kernel<<<B_ * H_, 256, 0, stream>>>(Rb, Db, Kf, Vr, Aa, Bb, state, Ob, O3);

    bonus_gate_kernel<<<M_, 64, 0, stream>>>(Ob, Rb, Kf, Vr, r_k, Gt, Xg);

    gemm_bf16_wmma<<<dim3(1024 / 64, M_ / 128), blk, 0, stream>>>(Xg, D_, Ow, 1024, Ou, 1024, D_, nullptr, 0, 0);

    residual_norm_kernel<<<M_, 256, 0, stream>>>(x_in, Ou, ln2, O1, O6);
}
